// MultiHeadAttention_70901320122713
// MI455X (gfx1250) — matrix operand order repaired and checked by reading, not compiled, not run on hardware
//
#include <hip/hip_runtime.h>

#define D_MODEL 1024
#define SEQ     2048
#define BATCH   2
#define NHEAD   16
#define HDIM    64
#define MROWS   (BATCH * SEQ)   // 4096

typedef __bf16 bf16;
typedef __attribute__((ext_vector_type(16))) __bf16 v16bf;
typedef __attribute__((ext_vector_type(8)))  __bf16 v8bf;
typedef __attribute__((ext_vector_type(4)))  __bf16 v4bf;
typedef __attribute__((ext_vector_type(8)))  float  v8f;
typedef int          v4i_g __attribute__((vector_size(16)));
typedef unsigned int v4u_g __attribute__((vector_size(16)));
typedef int          v8i_g __attribute__((vector_size(32)));

union Frag16 { v16bf v; v4bf q[4]; };
union Frag8f { v8f  v; float e[8]; };

// ---- CDNA5 async global->LDS copy (ASYNCcnt-tracked), guarded -------------
#if defined(__gfx1250__) &&                                                    \
    __has_builtin(__builtin_amdgcn_global_load_async_to_lds_b128) &&           \
    __has_builtin(__builtin_amdgcn_s_wait_asynccnt)
#define HAVE_ASYNC 1
#else
#define HAVE_ASYNC 0
#endif

// ---- CDNA5 Tensor Data Mover (TENSORcnt-tracked), guarded -----------------
#if defined(__gfx1250__) &&                                                    \
    __has_builtin(__builtin_amdgcn_tensor_load_to_lds) &&                      \
    __has_builtin(__builtin_amdgcn_s_wait_tensorcnt)
#define HAVE_TDM 1
#else
#define HAVE_TDM 0
#endif

__device__ __forceinline__ void cp16(bf16* dst_lds, const bf16* src_g) {
#if HAVE_ASYNC
  __builtin_amdgcn_global_load_async_to_lds_b128(
      (__attribute__((address_space(1))) v4i_g*)(
          __attribute__((address_space(1))) void*)(void*)src_g,
      (__attribute__((address_space(3))) v4i_g*)(
          __attribute__((address_space(3))) void*)(void*)dst_lds,
      0, 0);
#else
  *(v8bf*)dst_lds = *(const v8bf*)src_g;
#endif
}

#if HAVE_TDM
__device__ __forceinline__ unsigned lds_off(const void* p) {
  return (unsigned)(unsigned long long)(
      __attribute__((address_space(3))) char*)(void*)p;
}

// 2D TDM tile load, D# per ISA 8.3-8.6. Element = 2 bytes (bf16).
// LDS rows padded: 32 DWORDs (128B tile row) then +4 DWORDs (16B) -> 144B pitch.
__device__ __forceinline__ void tdm_load_2d(unsigned lds_addr, const void* gsrc,
                                            unsigned tensor_w, unsigned tensor_h,
                                            unsigned stride_elems,
                                            unsigned tile_w, unsigned tile_h) {
  unsigned long long ga = (unsigned long long)gsrc;
  v4u_g g0 = {1u,                                     // count=1, user flags 0
              lds_addr,                               // lds_addr[31:0]
              (unsigned)ga,                           // global_addr[31:0]
              (unsigned)((ga >> 32) & 0x1FFFFFFu) | (2u << 30)};  // [56:32]|type=2
  int d0 = (1 << 16)      // data_size = 2 bytes
         | (1 << 20)      // pad_enable
         | (4 << 22)      // pad_interval: 2^(4+1)=32 DWORDs (one 128B row)
         | (3 << 25);     // pad_amount: 4 DWORDs (16B)
  v8i_g g1 = {d0,
              (int)((tensor_w & 0xFFFFu) << 16),                       // dim0 lo
              (int)((tensor_w >> 16) | ((tensor_h & 0xFFFFu) << 16)),  // dim0 hi|dim1 lo
              (int)((tensor_h >> 16) | (tile_w << 16)),                // dim1 hi|tile0
              (int)tile_h,                                             // tile1 (tile2=0)
              (int)stride_elems,                                       // dim0_stride lo
              0, 0};
  v4i_g g2 = {1, 1, 0, 0};                    // tensor_dim2=1, tensor_dim3=1
  v4i_g g3 = {0, (int)(1u << 16), 0, 0};      // tensor_dim4=1
  v8i_g g4 = {0, 0, 0, 0, 0, 0, 0, 0};        // 5th group: zero-filled
  __builtin_amdgcn_tensor_load_to_lds(g0, g1, g2, g3, g4, 0);
}
#endif

__device__ __forceinline__ bf16 f2bf(float f) {
  unsigned u = __builtin_bit_cast(unsigned, f);
  u += 0x7fffu + ((u >> 16) & 1u);          // round-to-nearest-even
  unsigned short h = (unsigned short)(u >> 16);
  return __builtin_bit_cast(bf16, h);
}

// A-fragment (16x32 bf16, MxK): lane row = l%16, hi = l/16.
__device__ __forceinline__ Frag16 load_a16(const bf16* rowbase, int hi) {
  Frag16 f;
  const bf16* p0 = rowbase + 16 * hi;
  const bf16* p1 = rowbase + 16 * hi + 8;
  f.q[0] = *(const v4bf*)(p0);
  f.q[1] = *(const v4bf*)(p0 + 4);
  f.q[2] = *(const v4bf*)(p1);
  f.q[3] = *(const v4bf*)(p1 + 4);
  return f;
}

// B-fragment (32x16 bf16, KxN): lane col = l%16; p already includes 16*hi.
__device__ __forceinline__ Frag16 load_b16(const bf16* p) {
  Frag16 f;
  f.q[0] = ((const v4bf*)p)[0];
  f.q[1] = ((const v4bf*)p)[1];
  f.q[2] = ((const v4bf*)p)[2];
  f.q[3] = ((const v4bf*)p)[3];
  return f;
}

__device__ __forceinline__ v8f wmma_bf16(v16bf a, v16bf b, v8f c) {
  return __builtin_amdgcn_wmma_f32_16x16x32_bf16(false, a, false, b, (short)0, c,
                                                 false, false);
}

// ---------------------------------------------------------------------------
// fp32 -> bf16 conversion, 4 elements/thread
// ---------------------------------------------------------------------------
__global__ __launch_bounds__(256) void cvt_f32_bf16(const float* __restrict__ in,
                                                    bf16* __restrict__ out, int n4) {
  int i = blockIdx.x * 256 + threadIdx.x;
  if (i < n4) {
    float4 f = ((const float4*)in)[i];
    v4bf o;
    o[0] = f2bf(f.x); o[1] = f2bf(f.y); o[2] = f2bf(f.z); o[3] = f2bf(f.w);
    ((v4bf*)out)[i] = o;
  }
}

// ---------------------------------------------------------------------------
// Tiled bf16 WMMA GEMM: C[4096,1024] = A[M,K] * W[N,K]^T + bias
// 128x128 block tile, BK=32, double-buffered async LDS staging.
// MODE 0: bf16 [B,H,S,Dh] | MODE 1: bf16 [B,H,Dh,S] | MODE 2: fp32 [M,N]
// ---------------------------------------------------------------------------
template <int MODE>
__global__ __launch_bounds__(256) void gemm_bf16(const bf16* __restrict__ A,
                                                 const bf16* __restrict__ W,
                                                 const float* __restrict__ bias,
                                                 void* __restrict__ Out) {
  constexpr int K = D_MODEL;
  constexpr int PITCH = 40;                 // 80B rows (16B-aligned, low-conflict)
  constexpr int NK = K / 32;
  __shared__ __align__(16) bf16 sA[2][128 * PITCH];
  __shared__ __align__(16) bf16 sB[2][128 * PITCH];

  const int tid  = threadIdx.x;
  const int lane = tid & 31;
  const int wave = tid >> 5;
  const int hi   = lane >> 4;
  const int lr   = lane & 15;
  const int wm   = wave & 1;
  const int wn   = wave >> 1;
  const int m0   = blockIdx.y * 128;
  const int n0   = blockIdx.x * 128;

  // per tile: each wave issues exactly 4 async-copy instructions
  auto stage = [&](int buf, int k0) {
    for (int t = 0; t < 2; ++t) {
      int idx = tid + t * 256;              // 0..511
      int row = idx >> 2;                   // 0..127
      int c   = (idx & 3) << 3;             // 0,8,16,24 bf16 (16B chunks)
      cp16(&sA[buf][row * PITCH + c], &A[(size_t)(m0 + row) * K + k0 + c]);
      cp16(&sB[buf][row * PITCH + c], &W[(size_t)(n0 + row) * K + k0 + c]);
    }
  };

  Frag8f acc[4][2];
  for (int i = 0; i < 4; ++i)
    for (int j = 0; j < 2; ++j)
      for (int e = 0; e < 8; ++e) acc[i][j].e[e] = 0.f;

  stage(0, 0);
  for (int kt = 0; kt < NK; ++kt) {
    if (kt + 1 < NK) {
      stage((kt + 1) & 1, (kt + 1) * 32);
#if HAVE_ASYNC
      __builtin_amdgcn_s_wait_asynccnt(4);  // tile kt landed (in-order)
#endif
    } else {
#if HAVE_ASYNC
      __builtin_amdgcn_s_wait_asynccnt(0);
#endif
    }
    __syncthreads();

    const bf16* cA = sA[kt & 1];
    const bf16* cB = sB[kt & 1];
    Frag16 af[4], bfg[2];
    for (int i = 0; i < 4; ++i)
      af[i] = load_a16(&cA[(wm * 64 + i * 16 + lr) * PITCH], hi);
    for (int j = 0; j < 2; ++j)
      bfg[j] = load_b16(&cB[(wn * 32 + j * 16 + lr) * PITCH + 16 * hi]);

    for (int i = 0; i < 4; ++i)
      for (int j = 0; j < 2; ++j)
        acc[i][j].v = wmma_bf16(af[i].v, bfg[j].v, acc[i][j].v);
    __syncthreads();                        // protect buffer reuse at distance 2
  }

  for (int i = 0; i < 4; ++i)
    for (int j = 0; j < 2; ++j)
      for (int r = 0; r < 8; ++r) {
        int rg = m0 + wm * 64 + i * 16 + hi * 8 + r;
        int cg = n0 + wn * 32 + j * 16 + lr;
        float val = acc[i][j].e[r] + bias[cg];
        if (MODE == 2) {
          ((float*)Out)[(size_t)rg * D_MODEL + cg] = val;
        } else {
          int b = rg >> 11, s = rg & (SEQ - 1);
          int h = cg >> 6, d = cg & (HDIM - 1);
          bf16* O = (bf16*)Out;
          if (MODE == 0)
            O[((size_t)(b * NHEAD + h) * SEQ + s) * HDIM + d] = f2bf(val);
          else
            O[((size_t)(b * NHEAD + h) * HDIM + d) * SEQ + s] = f2bf(val);
        }
      }
}

// ---------------------------------------------------------------------------
// Flash attention: block = (b*H, 128 q-rows); 8 waves x 16 q-rows.
// K [64xDh] and Vt [Dhx64] chunks staged by the Tensor Data Mover (one
// wave-level DMA per tile, issued by wave 0, TENSORcnt-pipelined), with an
// async-LDS fallback. Double-buffered.
// ---------------------------------------------------------------------------
__global__ __launch_bounds__(256) void attn_fwd(const bf16* __restrict__ Q,
                                                const bf16* __restrict__ Kh_,
                                                const bf16* __restrict__ Vt_,
                                                bf16* __restrict__ AO) {
  constexpr int CP = 72;                    // 144B rows (16B-aligned)
  constexpr int NC = SEQ / 64;
  __shared__ __align__(16) bf16 sK[2][64 * CP];
  __shared__ __align__(16) bf16 sV[2][64 * CP];
  __shared__ __align__(16) bf16 sP[8][16 * CP];

  const int tid  = threadIdx.x;
  const int lane = tid & 31;
  const int wave = tid >> 5;
  const int hi   = lane >> 4;
  const int lr   = lane & 15;
  const int bh   = blockIdx.x;
  const int q0   = blockIdx.y * 128;

  const bf16* Qh = Q   + (size_t)bh * SEQ * HDIM;
  const bf16* Kh = Kh_ + (size_t)bh * SEQ * HDIM;
  const bf16* Vh = Vt_ + (size_t)bh * HDIM * SEQ;   // [d][s]

#if HAVE_TDM
  // two TDM descriptors per chunk; only called from wave 0
  auto stageT = [&](int buf, int c0) {
    tdm_load_2d(lds_off(&sK[buf][0]), Kh + (size_t)c0 * HDIM,
                HDIM, SEQ, HDIM, HDIM, 64);
    tdm_load_2d(lds_off(&sV[buf][0]), Vh + c0,
                SEQ, HDIM, SEQ, 64, 64);
  };
#else
  auto stageC = [&](int buf, int c0) {
    for (int t = 0; t < 2; ++t) {
      int idx = tid + t * 256;              // 0..511
      int row = idx >> 3;                   // 0..63
      int c   = (idx & 7) << 3;             // 0..56 bf16 (16B chunks)
      cp16(&sK[buf][row * CP + c], &Kh[(size_t)(c0 + row) * HDIM + c]);
      cp16(&sV[buf][row * CP + c], &Vh[(size_t)row * SEQ + c0 + c]);
    }
  };
#endif

  // resident Q fragments (16 rows x 64 d -> two 16x32 A-frags)
  const int qr = q0 + wave * 16 + lr;
  Frag16 qf[2];
  qf[0] = load_a16(Qh + (size_t)qr * HDIM, hi);
  qf[1] = load_a16(Qh + (size_t)qr * HDIM + 32, hi);

  Frag8f o[4];
  for (int jo = 0; jo < 4; ++jo)
    for (int e = 0; e < 8; ++e) o[jo].e[e] = 0.f;
  float mst[8], lst[8];
  for (int r = 0; r < 8; ++r) { mst[r] = -__builtin_inff(); lst[r] = 0.f; }

  constexpr float SCL = 0.125f * 1.44269504088896340736f;  // 1/sqrt(64)*log2(e)

#if HAVE_TDM
  if (wave == 0) stageT(0, 0);
#else
  stageC(0, 0);
#endif
  for (int ct = 0; ct < NC; ++ct) {
#if HAVE_TDM
    if (wave == 0) {
      if (ct + 1 < NC) {
        stageT((ct + 1) & 1, (ct + 1) * 64);
        __builtin_amdgcn_s_wait_tensorcnt(2);   // chunk ct landed (in-order)
      } else {
        __builtin_amdgcn_s_wait_tensorcnt(0);
      }
    }
#else
    if (ct + 1 < NC) {
      stageC((ct + 1) & 1, (ct + 1) * 64);
#if HAVE_ASYNC
      __builtin_amdgcn_s_wait_asynccnt(4);
#endif
    } else {
#if HAVE_ASYNC
      __builtin_amdgcn_s_wait_asynccnt(0);
#endif
    }
#endif
    __syncthreads();
    const bf16* cK = sK[ct & 1];
    const bf16* cV = sV[ct & 1];

    // scores: S(16x64) = Q(16x64) . Kchunk^T
    Frag8f sc[4];
    for (int jn = 0; jn < 4; ++jn) {
      for (int e = 0; e < 8; ++e) sc[jn].e[e] = 0.f;
      for (int kf = 0; kf < 2; ++kf) {
        Frag16 bk = load_b16(&cK[(jn * 16 + lr) * CP + 32 * kf + 16 * hi]);
        sc[jn].v = wmma_bf16(qf[kf].v, bk.v, sc[jn].v);
      }
      for (int e = 0; e < 8; ++e) sc[jn].e[e] *= SCL;
    }

    // online softmax (16-lane halves match C-frag row split)
    for (int r = 0; r < 8; ++r) {
      float mx = fmaxf(fmaxf(sc[0].e[r], sc[1].e[r]), fmaxf(sc[2].e[r], sc[3].e[r]));
      for (int off = 8; off >= 1; off >>= 1) mx = fmaxf(mx, __shfl_xor(mx, off, 32));
      float mn = fmaxf(mst[r], mx);
      float alpha = exp2f(mst[r] - mn);
      mst[r] = mn;
      float ps = 0.f;
      for (int jn = 0; jn < 4; ++jn) {
        float p = exp2f(sc[jn].e[r] - mn);
        sc[jn].e[r] = p;
        ps += p;
      }
      for (int off = 8; off >= 1; off >>= 1) ps += __shfl_xor(ps, off, 32);
      lst[r] = lst[r] * alpha + ps;
      for (int jo = 0; jo < 4; ++jo) o[jo].e[r] *= alpha;
    }

    // P (C-layout) -> LDS -> A-layout fragments
    for (int jn = 0; jn < 4; ++jn)
      for (int r = 0; r < 8; ++r)
        sP[wave][(8 * hi + r) * CP + jn * 16 + lr] = f2bf(sc[jn].e[r]);

    // O += P(16x64) . Vchunk(64x64)
    for (int kf = 0; kf < 2; ++kf) {
      Frag16 pa = load_a16(&sP[wave][lr * CP + 32 * kf], hi);
      for (int jo = 0; jo < 4; ++jo) {
        Frag16 bv = load_b16(&cV[(jo * 16 + lr) * CP + 32 * kf + 16 * hi]);
        o[jo].v = wmma_bf16(pa.v, bv.v, o[jo].v);
      }
    }
    __syncthreads();
  }

  const int b = bh >> 4, h = bh & 15;
  for (int r = 0; r < 8; ++r) {
    float inv = 1.f / lst[r];
    int s = q0 + wave * 16 + 8 * hi + r;
    size_t rowoff = ((size_t)(b * SEQ + s)) * D_MODEL + h * HDIM;
    for (int jo = 0; jo < 4; ++jo)
      AO[rowoff + jo * 16 + lr] = f2bf(o[jo].e[r] * inv);
  }
}

// ---------------------------------------------------------------------------
// Launch pipeline. ws layout: [0,8M) x_bf16 (reused as attn-out),
// [8M..16M) W_{q,k,v,o} bf16, [16M) Q, [24M) K, [32M) Vt -> 40 MB total.
// ---------------------------------------------------------------------------
extern "C" void kernel_launch(void* const* d_in, const int* in_sizes, int n_in,
                              void* d_out, int out_size, void* d_ws, size_t ws_size,
                              hipStream_t stream) {
  (void)in_sizes; (void)n_in; (void)out_size; (void)ws_size;
  const float* x   = (const float*)d_in[0];
  const float* W_q = (const float*)d_in[1];
  const float* b_q = (const float*)d_in[2];
  const float* W_k = (const float*)d_in[3];
  const float* b_k = (const float*)d_in[4];
  const float* W_v = (const float*)d_in[5];
  const float* b_v = (const float*)d_in[6];
  const float* W_o = (const float*)d_in[7];
  const float* b_o = (const float*)d_in[8];

  char* ws = (char*)d_ws;
  const size_t MB = 1u << 20;
  bf16* xb  = (bf16*)(ws + 0 * MB);
  bf16* wqb = (bf16*)(ws + 8 * MB);
  bf16* wkb = (bf16*)(ws + 10 * MB);
  bf16* wvb = (bf16*)(ws + 12 * MB);
  bf16* wob = (bf16*)(ws + 14 * MB);
  bf16* Qb  = (bf16*)(ws + 16 * MB);
  bf16* Kb  = (bf16*)(ws + 24 * MB);
  bf16* Vtb = (bf16*)(ws + 32 * MB);
  bf16* AOb = (bf16*)(ws + 0 * MB);         // reuse x_bf16 region (dead by then)

  const int n4x = MROWS * D_MODEL / 4;
  const int n4w = D_MODEL * D_MODEL / 4;
  cvt_f32_bf16<<<n4x / 256, 256, 0, stream>>>(x, xb, n4x);
  cvt_f32_bf16<<<n4w / 256, 256, 0, stream>>>(W_q, wqb, n4w);
  cvt_f32_bf16<<<n4w / 256, 256, 0, stream>>>(W_k, wkb, n4w);
  cvt_f32_bf16<<<n4w / 256, 256, 0, stream>>>(W_v, wvb, n4w);
  cvt_f32_bf16<<<n4w / 256, 256, 0, stream>>>(W_o, wob, n4w);

  dim3 ggrid(D_MODEL / 128, MROWS / 128);   // (8, 32)
  gemm_bf16<0><<<ggrid, 256, 0, stream>>>(xb, wqb, b_q, Qb);
  gemm_bf16<0><<<ggrid, 256, 0, stream>>>(xb, wkb, b_k, Kb);
  gemm_bf16<1><<<ggrid, 256, 0, stream>>>(xb, wvb, b_v, Vtb);

  attn_fwd<<<dim3(BATCH * NHEAD, SEQ / 128), 256, 0, stream>>>(Qb, Kb, Vtb, AOb);

  gemm_bf16<2><<<ggrid, 256, 0, stream>>>(AOb, wob, b_o, d_out);
}
